// EvidentialGNN3D_38208029065919
// MI455X (gfx1250) — matrix operand order repaired and checked by reading, not compiled, not run on hardware
//
#include <hip/hip_runtime.h>
#include <math.h>

#define SD 10
#define FEAT 14   // SD + 4 geo features

typedef _Float16 v16h __attribute__((ext_vector_type(16)));
typedef float    v8f  __attribute__((ext_vector_type(8)));
typedef _Float16 h4   __attribute__((ext_vector_type(4)));

union V16 { v16h v; _Float16 a[16]; };

static __device__ __forceinline__ void atomAddF(float* p, float v) {
    __hip_atomic_fetch_add(p, v, __ATOMIC_RELAXED, __HIP_MEMORY_SCOPE_AGENT);
}

// Branchless tanh: sign(x) * (1-u)/(1+u), u = exp2(-2*log2(e)*|x|).
// One v_exp_f32 + one v_rcp_f32, no exec-mask churn, no overflow (u <= 1).
static __device__ __forceinline__ float fast_tanh(float x) {
    float ax = fabsf(x);
    float u  = __builtin_amdgcn_exp2f(-2.885390081777927f * ax);
    float t  = (1.0f - u) * __builtin_amdgcn_rcpf(1.0f + u);
    return copysignf(t, x);
}

// ---------------------------------------------------------------------------
// Kernel 1: per-edge geometric features, stored packed as f16x4 (8B/edge).
// geo = [edge_len, sum|c_from|, dot(c_from,c_to), dot(c_from,edge_vec)]
// ---------------------------------------------------------------------------
__global__ __launch_bounds__(256) void geo_kernel(
    const float* __restrict__ coords, const float* __restrict__ elen,
    const float* __restrict__ evec, const int* __restrict__ nfrom,
    const int* __restrict__ nto, h4* __restrict__ geo, int E_)
{
    int e = blockIdx.x * blockDim.x + threadIdx.x;
    if (e >= E_) return;
    int f = nfrom[e], t = nto[e];
    float fx = coords[3*f+0], fy = coords[3*f+1], fz = coords[3*f+2];
    float tx = coords[3*t+0], ty = coords[3*t+1], tz = coords[3*t+2];
    float vx = evec[3*e+0],  vy = evec[3*e+1],  vz = evec[3*e+2];
    float cs = fabsf(fx) + fabsf(fy) + fabsf(fz);
    float d1 = fx*tx + fy*ty + fz*tz;
    float d2 = fx*vx + fy*vy + fz*vz;
    h4 g;
    g[0] = (_Float16)elen[e];
    g[1] = (_Float16)cs;
    g[2] = (_Float16)d1;
    g[3] = (_Float16)d2;
    geo[e] = g;
}

// ---------------------------------------------------------------------------
// Kernel 2: one wave handles a tile of 16 edges.
//   A (16x32 f16): row m = edge tileBase+m, K0..9 = state[node_from], K10..13
//                  = geo, rest zero-padded.
//   B (32x16 f16): column n = W_msg[:,n] zero-padded (built once per wave).
//   C (f32):       per-column bias b_msg[n] (bias folded into WMMA accum).
//   D = A x B + C via v_wmma_f32_16x16x32_f16, then branchless tanh +
//   coalesced global f32 atomic scatter into state_next[node_to].
// ---------------------------------------------------------------------------
__global__ __launch_bounds__(256) void msg_kernel(
    const float* __restrict__ state, float* __restrict__ state_next,
    const h4* __restrict__ geo, const int* __restrict__ nfrom,
    const int* __restrict__ nto, const float* __restrict__ W_msg,
    const float* __restrict__ b_msg, int E_)
{
    const int lane     = threadIdx.x & 31;
    const int m        = lane & 15;        // A row / B-C-D column index
    const int half_sel = lane >> 4;        // 0: K 0-7 & 16-23 ; 1: K 8-15 & 24-31
    const int wpb      = blockDim.x >> 5;
    int wave  = blockIdx.x * wpb + (threadIdx.x >> 5);
    int step  = gridDim.x * wpb;
    int tiles = E_ >> 4;

    // Build B operand once (constant across tiles). Column n = lane%16.
    const int n = m;
    V16 B;
#pragma unroll
    for (int j = 0; j < 16; ++j) {
        int K = (j < 8) ? half_sel * 8 + j : 16 + half_sel * 8 + (j - 8);
        float w = (K < FEAT && n < SD) ? W_msg[K * SD + n] : 0.0f;
        B.a[j] = (_Float16)w;
    }
    const float bias = (n < SD) ? b_msg[n] : 0.0f;

    for (int t = wave; t < tiles; t += step) {
        const int e  = (t << 4) + m;
        const int nf = nfrom[e];
        const float* __restrict__ srow = state + nf * SD;

        V16 A;
#pragma unroll
        for (int j = 0; j < 16; ++j) A.a[j] = (_Float16)0.0f;
        if (half_sel == 0) {
            // elements 0..7 -> K=0..7 (state dims 0..7); 8..15 -> K=16..23 (pad)
#pragma unroll
            for (int j = 0; j < 8; ++j) A.a[j] = (_Float16)srow[j];
        } else {
            // elements 0..7 -> K=8..15: state8, state9, geo0..3, pad, pad
            A.a[0] = (_Float16)srow[8];
            A.a[1] = (_Float16)srow[9];
            h4 g = geo[e];
            A.a[2] = g[0]; A.a[3] = g[1]; A.a[4] = g[2]; A.a[5] = g[3];
        }

        // Bias in the accumulator: lane owns column n for rows 8*half_sel+r.
        v8f C;
#pragma unroll
        for (int r = 0; r < 8; ++r) C[r] = bias;

        C = __builtin_amdgcn_wmma_f32_16x16x32_f16(
                /*neg_a=*/false, A.v, /*neg_b=*/false, B.v,
                /*c_mod=*/(short)0, C, /*reuse_a=*/false, /*reuse_b=*/false);

        // D layout: vgpr r, this lane -> (M = half_sel*8 + r, N = n)
        if (n < SD) {
            // 8 destination indices for this lane's rows (32B-aligned tile).
            const int4* nt4 = (const int4*)(nto + (t << 4) + half_sel * 8);
            int4 d0 = nt4[0];
            int4 d1 = nt4[1];
            int dst[8] = { d0.x, d0.y, d0.z, d0.w, d1.x, d1.y, d1.z, d1.w };
#pragma unroll
            for (int r = 0; r < 8; ++r) {
                // lanes 0..9 of each half hit 10 consecutive floats: coalesced
                atomAddF(&state_next[dst[r] * SD + n], fast_tanh(C[r]));
            }
        }
    }
}

// Scalar fallback for E % 16 leftover edges (not hit for E = 6.4M).
__global__ __launch_bounds__(256) void msg_tail_kernel(
    const float* __restrict__ state, float* __restrict__ state_next,
    const h4* __restrict__ geo, const int* __restrict__ nfrom,
    const int* __restrict__ nto, const float* __restrict__ W_msg,
    const float* __restrict__ b_msg, int start, int E_)
{
    int e = start + blockIdx.x * blockDim.x + threadIdx.x;
    if (e >= E_) return;
    float inp[FEAT];
    const float* s = state + nfrom[e] * SD;
#pragma unroll
    for (int k = 0; k < SD; ++k) inp[k] = s[k];
    h4 g = geo[e];
    inp[10] = (float)g[0]; inp[11] = (float)g[1];
    inp[12] = (float)g[2]; inp[13] = (float)g[3];
    int dst = nto[e];
#pragma unroll
    for (int c = 0; c < SD; ++c) {
        float acc = b_msg[c];
#pragma unroll
        for (int k = 0; k < FEAT; ++k) acc += inp[k] * W_msg[k * SD + c];
        atomAddF(&state_next[dst * SD + c], fast_tanh(acc));
    }
}

// ---------------------------------------------------------------------------
// Kernel 3: per-graph reduction of node states (graph accumulator is 80 KB,
// L2-resident; plain f32 atomics).
// ---------------------------------------------------------------------------
__global__ __launch_bounds__(256) void graph_accum_kernel(
    const float* __restrict__ state, const int* __restrict__ ngi,
    float* __restrict__ gstate, int N_)
{
    int i = blockIdx.x * blockDim.x + threadIdx.x;
    if (i >= N_) return;
    int g = ngi[i];
#pragma unroll
    for (int c = 0; c < SD; ++c)
        atomAddF(&gstate[g * SD + c], state[i * SD + c]);
}

// ---------------------------------------------------------------------------
// Kernel 4: (G,10) @ (10,4) + bias, then evidential head transforms.
// ---------------------------------------------------------------------------
static __device__ __forceinline__ float softplus_f(float x) {
    return fmaxf(x, 0.0f) + log1pf(expf(-fabsf(x)));
}

__global__ __launch_bounds__(256) void out_kernel(
    const float* __restrict__ gstate, const float* __restrict__ W_out,
    const float* __restrict__ b_out, float* __restrict__ out, int G_)
{
    int g = blockIdx.x * blockDim.x + threadIdx.x;
    if (g >= G_) return;
    float ev[4];
#pragma unroll
    for (int j = 0; j < 4; ++j) {
        float s = b_out[j];
#pragma unroll
        for (int c = 0; c < SD; ++c) s += gstate[g * SD + c] * W_out[c * 4 + j];
        ev[j] = s;
    }
    out[g * 4 + 0] = ev[0];
    out[g * 4 + 1] = softplus_f(ev[1]);
    out[g * 4 + 2] = softplus_f(ev[2]) + 1.0f;
    out[g * 4 + 3] = softplus_f(ev[3]);
}

// ---------------------------------------------------------------------------
extern "C" void kernel_launch(void* const* d_in, const int* in_sizes, int n_in,
                              void* d_out, int out_size, void* d_ws, size_t ws_size,
                              hipStream_t stream)
{
    const float* coords = (const float*)d_in[0];
    const float* elen   = (const float*)d_in[1];
    const float* evec   = (const float*)d_in[2];
    const float* W_msg  = (const float*)d_in[3];
    const float* b_msg  = (const float*)d_in[4];
    const float* W_out  = (const float*)d_in[5];
    const float* b_out  = (const float*)d_in[6];
    const int*   nfrom  = (const int*)d_in[7];
    const int*   nto    = (const int*)d_in[8];
    const int*   ngi    = (const int*)d_in[9];

    const int N_ = in_sizes[0] / 3;   // node_coordinates (N,3)
    const int E_ = in_sizes[1];       // edge_lengths (E,1)
    const int G_ = out_size / 4;      // output (G,4)

    // Workspace layout (256B aligned): geo f16x4 | stateA | stateB | gstate
    char*  ws  = (char*)d_ws;
    size_t off = 0;
    auto carve = [&](size_t bytes) -> char* {
        char* p = ws + off;
        off = (off + bytes + 255) & ~(size_t)255;
        return p;
    };
    h4*    geo    = (h4*)   carve((size_t)E_ * sizeof(h4));
    float* stateA = (float*)carve((size_t)N_ * SD * sizeof(float));
    float* stateB = (float*)carve((size_t)N_ * SD * sizeof(float));
    float* gstate = (float*)carve((size_t)G_ * SD * sizeof(float));
    (void)ws_size;

    // Geometric features (round-invariant).
    geo_kernel<<<(E_ + 255) / 256, 256, 0, stream>>>(coords, elen, evec,
                                                     nfrom, nto, geo, E_);

    // state = 0
    hipMemsetAsync(stateA, 0, (size_t)N_ * SD * sizeof(float), stream);
    hipMemsetAsync(gstate, 0, (size_t)G_ * SD * sizeof(float), stream);

    const int tiles    = E_ >> 4;
    const int tailBase = tiles << 4;
    const int tailCnt  = E_ - tailBase;
    const int wpb      = 8;                         // 256 threads = 8 waves
    const int msgBlks  = (tiles + wpb - 1) / wpb;

    float* cur = stateA;
    float* nxt = stateB;
    for (int r = 0; r < 3; ++r) {
        // state_next starts as a copy of state (state += segment_sum(msg)).
        hipMemcpyAsync(nxt, cur, (size_t)N_ * SD * sizeof(float),
                       hipMemcpyDeviceToDevice, stream);
        if (tiles > 0)
            msg_kernel<<<msgBlks, 256, 0, stream>>>(cur, nxt, geo, nfrom, nto,
                                                    W_msg, b_msg, E_);
        if (tailCnt > 0)
            msg_tail_kernel<<<(tailCnt + 255) / 256, 256, 0, stream>>>(
                cur, nxt, geo, nfrom, nto, W_msg, b_msg, tailBase, E_);
        float* tmp = cur; cur = nxt; nxt = tmp;
    }

    graph_accum_kernel<<<(N_ + 255) / 256, 256, 0, stream>>>(cur, ngi, gstate, N_);
    out_kernel<<<(G_ + 255) / 256, 256, 0, stream>>>(gstate, W_out, b_out,
                                                     (float*)d_out, G_);
}
